// HiLo_3788161155364
// MI455X (gfx1250) — matrix operand order repaired and checked by reading, not compiled, not run on hardware
//
#include <hip/hip_runtime.h>
#include <hip/hip_bf16.h>
#include <hip/hip_fp16.h>

// ---- problem constants (fixed by setup_inputs) ----
#define BATCH   8
#define NTOK    4096        // 64*64
#define DIMC    512
#define KVLEN   1024        // 32*32 pooled
#define SCALE_F 0.125f      // 64^-0.5

typedef __attribute__((ext_vector_type(16))) _Float16 v16h;
typedef __attribute__((ext_vector_type(8)))  _Float16 v8h;
typedef __attribute__((ext_vector_type(8)))  float    v8f;

union Frag16 { v16h v; v8h h[2]; };

static __device__ inline v16h ld_frag(const _Float16* p0, const _Float16* p1) {
  Frag16 f;
  f.h[0] = *(const v8h*)p0;
  f.h[1] = *(const v8h*)p1;
  return f.v;
}

static __device__ inline v8f wmma_f16(v16h a, v16h b, v8f c) {
  return __builtin_amdgcn_wmma_f32_16x16x32_f16(false, a, false, b, (short)0, c,
                                                false, false);
}

static __device__ inline float red_sum32(float v) {
  v += __shfl_xor(v, 1, 32);  v += __shfl_xor(v, 2, 32);
  v += __shfl_xor(v, 4, 32);  v += __shfl_xor(v, 8, 32);
  v += __shfl_xor(v, 16, 32); return v;
}
static __device__ inline float row_max16(float v) {
  v = fmaxf(v, __shfl_xor(v, 1, 32)); v = fmaxf(v, __shfl_xor(v, 2, 32));
  v = fmaxf(v, __shfl_xor(v, 4, 32)); v = fmaxf(v, __shfl_xor(v, 8, 32));
  return v;
}
static __device__ inline float row_sum16(float v) {
  v += __shfl_xor(v, 1, 32); v += __shfl_xor(v, 2, 32);
  v += __shfl_xor(v, 4, 32); v += __shfl_xor(v, 8, 32);
  return v;
}

// ---------------- elementwise / prep kernels ----------------
__global__ void k_cvt_f16(const float* __restrict__ x, _Float16* __restrict__ y, int n) {
  int i = blockIdx.x * blockDim.x + threadIdx.x;
  if (i < n) y[i] = (_Float16)x[i];
}

// pooled[b][pg][c] = mean over 2x2 window of x4
__global__ void k_pool(const float* __restrict__ x, _Float16* __restrict__ pooled) {
  int i = blockIdx.x * blockDim.x + threadIdx.x;           // < B*1024*512
  int ch = i & 511;
  int pg = (i >> 9) & 1023;
  int b  = i >> 19;
  int ph = pg >> 5, pw = pg & 31;
  int t0 = ph * 128 + pw * 2;                              // (ph*2)*64 + pw*2
  const float* p = x + ((size_t)b * NTOK + t0) * DIMC + ch;
  float s = p[0] + p[512] + p[64 * 512] + p[65 * 512];
  pooled[i] = (_Float16)(s * 0.25f);
}

// wT[n*K + k] = (f16) w[k*N + n]
__global__ void k_wconv(const float* __restrict__ w, _Float16* __restrict__ wt,
                        int K, int N) {
  int i = blockIdx.x * blockDim.x + threadIdx.x;
  if (i >= K * N) return;
  int n = i / K, k = i - n * K;
  wt[i] = (_Float16)w[(size_t)k * N + n];
}

// ---------------- WMMA GEMM core: one wave -> 64x64 tile ----------------
// 16 wmma per K-step vs 16 b128 loads per lane (1.0 wmma/load)
static __device__ inline void gemm_tile_64x64(const _Float16* __restrict__ A, int lda,
                                              const _Float16* __restrict__ BT, int ldb,
                                              int K, int m0, int n0, v8f acc[16]) {
  const int lane = threadIdx.x & 31;
  const int lm = lane & 15, g = lane >> 4;
  const _Float16* arow[4];
#pragma unroll
  for (int r = 0; r < 4; ++r)
    arow[r] = A + (size_t)(m0 + r * 16 + lm) * lda;
  const _Float16* brow[4];
#pragma unroll
  for (int t = 0; t < 4; ++t)
    brow[t] = BT + (size_t)(n0 + t * 16 + lm) * ldb;

  for (int k0 = 0; k0 < K; k0 += 32) {
    v16h bfr[4];
#pragma unroll
    for (int t = 0; t < 4; ++t)
      bfr[t] = ld_frag(brow[t] + k0 + g * 16, brow[t] + k0 + g * 16 + 8);
#pragma unroll
    for (int r = 0; r < 4; ++r) {
      v16h a = ld_frag(arow[r] + k0 + g * 16, arow[r] + k0 + g * 16 + 8);
#pragma unroll
      for (int t = 0; t < 4; ++t)
        acc[r * 4 + t] = wmma_f16(a, bfr[t], acc[r * 4 + t]);
    }
  }
}

// C (f16, row-major)
__global__ __launch_bounds__(32) void k_gemm_f16(const _Float16* __restrict__ A,
                                                 const _Float16* __restrict__ BT,
                                                 _Float16* __restrict__ C,
                                                 int lda, int ldb, int ldc, int K) {
  int m0 = blockIdx.x * 64, n0 = blockIdx.y * 64;
  v8f acc[16] = {};
  gemm_tile_64x64(A, lda, BT, ldb, K, m0, n0, acc);
  const int lane = threadIdx.x & 31, lm = lane & 15, g = lane >> 4;
#pragma unroll
  for (int r = 0; r < 4; ++r)
#pragma unroll
    for (int t = 0; t < 4; ++t)
#pragma unroll
      for (int v = 0; v < 8; ++v)
        C[(size_t)(m0 + r * 16 + v + 8 * g) * ldc + (n0 + t * 16 + lm)] =
            (_Float16)acc[r * 4 + t][v];
}

// KV projection: A = pooled [8192][512]; cols 0..255 -> K row-major, 256..511 -> V transposed [b][h][d][kv]
__global__ __launch_bounds__(32) void k_gemm_kv(const _Float16* __restrict__ A,
                                                const _Float16* __restrict__ BT,
                                                _Float16* __restrict__ klo,
                                                _Float16* __restrict__ vT) {
  int m0 = blockIdx.x * 64, n0 = blockIdx.y * 64;
  v8f acc[16] = {};
  gemm_tile_64x64(A, DIMC, BT, DIMC, DIMC, m0, n0, acc);
  const int lane = threadIdx.x & 31, lm = lane & 15, g = lane >> 4;
#pragma unroll
  for (int r = 0; r < 4; ++r)
#pragma unroll
    for (int t = 0; t < 4; ++t) {
      int col = n0 + t * 16 + lm;
#pragma unroll
      for (int v = 0; v < 8; ++v) {
        int row = m0 + r * 16 + v + 8 * g;
        _Float16 val = (_Float16)acc[r * 4 + t][v];
        if (col < 256) {
          klo[(size_t)row * 256 + col] = val;
        } else {
          int c = col - 256, h = c >> 6, d = c & 63;
          int b = row >> 10, kv = row & 1023;
          vT[(((size_t)b * 4 + h) * 64 + d) * KVLEN + kv] = val;
        }
      }
    }
}

// final projection: f32 out with bias, output row stride 512, column offset
__global__ __launch_bounds__(32) void k_gemm_bias(const _Float16* __restrict__ A,
                                                  const _Float16* __restrict__ BT,
                                                  const float* __restrict__ bias,
                                                  float* __restrict__ out, int coloff) {
  int m0 = blockIdx.x * 64, n0 = blockIdx.y * 64;
  v8f acc[16] = {};
  gemm_tile_64x64(A, 256, BT, 256, 256, m0, n0, acc);
  const int lane = threadIdx.x & 31, lm = lane & 15, g = lane >> 4;
#pragma unroll
  for (int r = 0; r < 4; ++r)
#pragma unroll
    for (int t = 0; t < 4; ++t) {
      int col = n0 + t * 16 + lm;
      float bv = bias[col];
#pragma unroll
      for (int v = 0; v < 8; ++v)
        out[(size_t)(m0 + r * 16 + v + 8 * g) * DIMC + coloff + col] =
            acc[r * 4 + t][v] + bv;
    }
}

// ---------------- HiFi: 2x2 windowed attention (VALU; ~0.1 GFLOP) ----------------
__global__ __launch_bounds__(128) void k_hifi(const _Float16* __restrict__ qkv,
                                              _Float16* __restrict__ outw) {
  int b = blockIdx.x >> 10, tg = blockIdx.x & 1023;
  int h = threadIdx.x >> 5, lane = threadIdx.x & 31;
  int d0 = lane * 2;
  int wh = tg >> 5, ww = tg & 31;
  int t0 = wh * 128 + ww * 2;
  int tok[4] = {t0, t0 + 1, t0 + 64, t0 + 65};
  float q[4][2], k[4][2], vv[4][2];
#pragma unroll
  for (int i = 0; i < 4; ++i) {
    const _Float16* p = qkv + ((size_t)(b * NTOK + tok[i])) * 768 + h * 64 + d0;
    q[i][0]  = (float)p[0];   q[i][1]  = (float)p[1];
    k[i][0]  = (float)p[256]; k[i][1]  = (float)p[257];
    vv[i][0] = (float)p[512]; vv[i][1] = (float)p[513];
  }
  float s[4][4];
#pragma unroll
  for (int i = 0; i < 4; ++i)
#pragma unroll
    for (int j = 0; j < 4; ++j)
      s[i][j] = red_sum32(q[i][0] * k[j][0] + q[i][1] * k[j][1]) * SCALE_F;
#pragma unroll
  for (int i = 0; i < 4; ++i) {
    float mx = fmaxf(fmaxf(s[i][0], s[i][1]), fmaxf(s[i][2], s[i][3]));
    float e0 = __expf(s[i][0] - mx), e1 = __expf(s[i][1] - mx);
    float e2 = __expf(s[i][2] - mx), e3 = __expf(s[i][3] - mx);
    float inv = 1.0f / (e0 + e1 + e2 + e3);
    float o0 = (e0 * vv[0][0] + e1 * vv[1][0] + e2 * vv[2][0] + e3 * vv[3][0]) * inv;
    float o1 = (e0 * vv[0][1] + e1 * vv[1][1] + e2 * vv[2][1] + e3 * vv[3][1]) * inv;
    _Float16* op = outw + ((size_t)(b * NTOK + tok[i])) * 256 + h * 64 + d0;
    op[0] = (_Float16)o0; op[1] = (_Float16)o1;
  }
}

// ---------------- LoFi: flash attention, 16 query rows per wave ----------------
__global__ __launch_bounds__(32) void k_lofi(const _Float16* __restrict__ qlo,
                                             const _Float16* __restrict__ klo,
                                             const _Float16* __restrict__ vT,
                                             _Float16* __restrict__ outw) {
  int qt = blockIdx.x & 255;
  int h  = (blockIdx.x >> 8) & 3;
  int b  = blockIdx.x >> 10;
  const int lane = threadIdx.x & 31, lm = lane & 15, g = lane >> 4;
  __shared__ _Float16 Pl[16 * 32];

  const _Float16* qrow = qlo + ((size_t)(b * NTOK + qt * 16 + lm)) * 256 + h * 64;
  v16h aq0 = ld_frag(qrow + g * 16,      qrow + g * 16 + 8);
  v16h aq1 = ld_frag(qrow + 32 + g * 16, qrow + 32 + g * 16 + 8);

  v8f acc[4] = {};
  float m[8], lsum[8];
#pragma unroll
  for (int v = 0; v < 8; ++v) { m[v] = -1e30f; lsum[v] = 0.0f; }

  for (int j = 0; j < KVLEN; j += 32) {
    v8f s0 = {}, s1 = {};
    {
      const _Float16* kp = klo + ((size_t)(b * KVLEN + j + lm)) * 256 + h * 64;
      v16h b0 = ld_frag(kp + g * 16,      kp + g * 16 + 8);
      v16h b1 = ld_frag(kp + 32 + g * 16, kp + 32 + g * 16 + 8);
      s0 = wmma_f16(aq0, b0, s0);
      s0 = wmma_f16(aq1, b1, s0);
    }
    {
      const _Float16* kp = klo + ((size_t)(b * KVLEN + j + 16 + lm)) * 256 + h * 64;
      v16h b0 = ld_frag(kp + g * 16,      kp + g * 16 + 8);
      v16h b1 = ld_frag(kp + 32 + g * 16, kp + 32 + g * 16 + 8);
      s1 = wmma_f16(aq0, b0, s1);
      s1 = wmma_f16(aq1, b1, s1);
    }
    float p0[8], p1[8], alpha[8];
#pragma unroll
    for (int v = 0; v < 8; ++v) {
      float a0 = s0[v] * SCALE_F, a1 = s1[v] * SCALE_F;
      float mx = row_max16(fmaxf(a0, a1));
      float mnew = fmaxf(m[v], mx);
      float al = __expf(m[v] - mnew);
      float e0 = __expf(a0 - mnew), e1 = __expf(a1 - mnew);
      float rs = row_sum16(e0 + e1);
      lsum[v] = lsum[v] * al + rs;
      m[v] = mnew; alpha[v] = al; p0[v] = e0; p1[v] = e1;
    }
#pragma unroll
    for (int t = 0; t < 4; ++t)
#pragma unroll
      for (int v = 0; v < 8; ++v) acc[t][v] *= alpha[v];
    // stage P (C-layout f32) -> LDS -> reload as A-layout f16
#pragma unroll
    for (int v = 0; v < 8; ++v) {
      int r = v + 8 * g;
      Pl[r * 32 + lm]      = (_Float16)p0[v];
      Pl[r * 32 + 16 + lm] = (_Float16)p1[v];
    }
    __syncthreads();
    v16h ap = ld_frag((const _Float16*)Pl + lm * 32 + g * 16,
                      (const _Float16*)Pl + lm * 32 + g * 16 + 8);
    __syncthreads();
#pragma unroll
    for (int t = 0; t < 4; ++t) {
      const _Float16* vp =
          vT + (((size_t)(b * 4 + h)) * 64 + t * 16 + lm) * KVLEN + j + g * 16;
      v16h bv = ld_frag(vp, vp + 8);
      acc[t] = wmma_f16(ap, bv, acc[t]);
    }
  }
#pragma unroll
  for (int t = 0; t < 4; ++t)
#pragma unroll
    for (int v = 0; v < 8; ++v) {
      int r = v + 8 * g;
      float val = acc[t][v] / lsum[v];
      outw[((size_t)(b * NTOK + qt * 16 + r)) * 256 + h * 64 + t * 16 + lm] =
          (_Float16)val;
    }
}

// ---------------- host launcher ----------------
extern "C" void kernel_launch(void* const* d_in, const int* in_sizes, int n_in,
                              void* d_out, int out_size, void* d_ws, size_t ws_size,
                              hipStream_t stream) {
  (void)in_sizes; (void)n_in; (void)out_size; (void)ws_size;
  const float* x     = (const float*)d_in[0];
  const float* lqw   = (const float*)d_in[3];   // [512,256]
  const float* lkvw  = (const float*)d_in[4];   // [512,512]
  const float* lpw   = (const float*)d_in[5];   // [256,256]
  const float* lpb   = (const float*)d_in[6];   // [256]
  const float* hqkvw = (const float*)d_in[7];   // [512,768]
  const float* hpw   = (const float*)d_in[8];   // [256,256]
  const float* hpb   = (const float*)d_in[9];   // [256]
  float* out = (float*)d_out;

  _Float16* p = (_Float16*)d_ws;
  _Float16* xb      = p; p += (size_t)BATCH * NTOK * DIMC;     // 16.8M
  _Float16* pooled  = p; p += (size_t)BATCH * KVLEN * DIMC;    // 4.2M
  _Float16* wT_lq   = p; p += 256 * 512;
  _Float16* wT_lkv  = p; p += 512 * 512;
  _Float16* wT_hq   = p; p += 768 * 512;
  _Float16* wT_lp   = p; p += 256 * 256;
  _Float16* wT_hp   = p; p += 256 * 256;
  _Float16* qlo     = p; p += (size_t)BATCH * NTOK * 256;      // 8.4M
  _Float16* klo     = p; p += (size_t)BATCH * KVLEN * 256;     // 2.1M
  _Float16* vT      = p; p += (size_t)BATCH * 4 * 64 * KVLEN;  // 2.1M
  _Float16* qkvh    = p; p += (size_t)BATCH * NTOK * 768;      // 25.2M
  // xb is dead after the three input GEMMs; reuse it for attention outputs
  _Float16* attn_hi = xb;
  _Float16* attn_lo = xb + (size_t)BATCH * NTOK * 256;

  const int M = BATCH * NTOK;                  // 32768

  k_cvt_f16<<<(M * DIMC + 255) / 256, 256, 0, stream>>>(x, xb, M * DIMC);
  k_pool<<<(BATCH * KVLEN * DIMC + 255) / 256, 256, 0, stream>>>(x, pooled);
  k_wconv<<<(512 * 256 + 255) / 256, 256, 0, stream>>>(lqw,   wT_lq,  512, 256);
  k_wconv<<<(512 * 512 + 255) / 256, 256, 0, stream>>>(lkvw,  wT_lkv, 512, 512);
  k_wconv<<<(512 * 768 + 255) / 256, 256, 0, stream>>>(hqkvw, wT_hq,  512, 768);
  k_wconv<<<(256 * 256 + 255) / 256, 256, 0, stream>>>(lpw,   wT_lp,  256, 256);
  k_wconv<<<(256 * 256 + 255) / 256, 256, 0, stream>>>(hpw,   wT_hp,  256, 256);

  // q_lo = xb @ l_q_w            [32768,512]x[512,256]
  k_gemm_f16<<<dim3(M / 64, 256 / 64), 32, 0, stream>>>(xb, wT_lq, qlo, DIMC, DIMC, 256, DIMC);
  // kv = pooled @ l_kv_w         [8192,512]x[512,512]  -> klo + vT
  k_gemm_kv<<<dim3(BATCH * KVLEN / 64, 512 / 64), 32, 0, stream>>>(pooled, wT_lkv, klo, vT);
  // qkv_hi = xb @ h_qkv_w        [32768,512]x[512,768]
  k_gemm_f16<<<dim3(M / 64, 768 / 64), 32, 0, stream>>>(xb, wT_hq, qkvh, DIMC, DIMC, 768, DIMC);

  k_hifi<<<BATCH * 1024, 128, 0, stream>>>(qkvh, attn_hi);
  k_lofi<<<BATCH * 4 * (NTOK / 16), 32, 0, stream>>>(qlo, klo, vT, attn_lo);

  // hifi_out -> channels [0,256), lofi_out -> channels [256,512)
  k_gemm_bias<<<dim3(M / 64, 256 / 64), 32, 0, stream>>>(attn_hi, wT_hp, hpb, out, 0);
  k_gemm_bias<<<dim3(M / 64, 256 / 64), 32, 0, stream>>>(attn_lo, wT_lp, lpb, out, 256);
}
